// DeformableConv2D_66846870995178
// MI455X (gfx1250) — matrix operand order repaired and checked by reading, not compiled, not run on hardware
//
#include <hip/hip_runtime.h>
#include <hip/hip_bf16.h>
#include <math.h>

typedef _Float16 v16h __attribute__((ext_vector_type(16)));
typedef _Float16 v8h  __attribute__((ext_vector_type(8)));
typedef float    v8f  __attribute__((ext_vector_type(8)));

#define B_ 8
#define C_ 256
#define H_ 64
#define W_ 64
#define F_ 256
#define K_ 9

// A-tile row stride in halfs: 264*2B = 528B (132 dwords ≡ 4 mod 64 banks)
// -> pixel rows map to disjoint LDS banks for b128 accesses.
#define AST 264

// ---------------------------------------------------------------------------
// Stage 0: repack w_deform[F, C, 3, 3] (f32) -> Wt[k][f][c] (f16), i.e.
// Wt[(k*F + f)*C + c]. A WMMA B-fragment (lane = column f, half = K-group)
// is then two runs of 8 contiguous f16 (16 apart) -> two 16B global loads, no LDS staging needed.
// ---------------------------------------------------------------------------
__global__ __launch_bounds__(256) void k_repack_w(const float* __restrict__ wd,
                                                  _Float16* __restrict__ wt) {
  int idx = blockIdx.x * 256 + threadIdx.x;     // over K*F*C = 589824
  if (idx >= K_ * F_ * C_) return;
  int c  = idx & (C_ - 1);
  int kf = idx >> 8;          // k*F + f
  int f  = kf & (F_ - 1);
  int k  = kf >> 8;
  wt[idx] = (_Float16)wd[(f * C_ + c) * K_ + k];
}

// ---------------------------------------------------------------------------
// Stage 1: 3x3 SAME convs producing 18 offset channels + 9 mask channels
// (sigmoid applied). One WG = 16x16 spatial tile of one image; loop over C
// with x halo tile (18x18) and per-channel 27x9 weights staged in LDS.
// ---------------------------------------------------------------------------
__global__ __launch_bounds__(256) void k_offset_mask(const float* __restrict__ x,
                                                     const float* __restrict__ w_off,
                                                     const float* __restrict__ w_msk,
                                                     float* __restrict__ off_ws,
                                                     float* __restrict__ msk_ws) {
  __shared__ float xt[18 * 18];
  __shared__ float wl[27 * 9];
  int t    = threadIdx.x;
  int b    = blockIdx.x >> 4;
  int tile = blockIdx.x & 15;
  int ty0  = (tile >> 2) * 16;
  int tx0  = (tile & 3) * 16;
  int py = t >> 4, px = t & 15;
  int y = ty0 + py, xw = tx0 + px;

  float acc[27];
#pragma unroll
  for (int i = 0; i < 27; ++i) acc[i] = 0.f;

  for (int c = 0; c < C_; ++c) {
    for (int idx = t; idx < 324; idx += 256) {
      int iy = idx / 18, ix = idx - iy * 18;
      int gy = ty0 - 1 + iy, gx = tx0 - 1 + ix;
      float v = 0.f;
      if (gy >= 0 && gy < H_ && gx >= 0 && gx < W_)
        v = x[((b * C_ + c) * H_ + gy) * W_ + gx];
      xt[idx] = v;
    }
    for (int idx = t; idx < 243; idx += 256) {
      int ch = idx / 9, kk = idx - ch * 9;
      wl[idx] = (ch < 18) ? w_off[(ch * C_ + c) * 9 + kk]
                          : w_msk[((ch - 18) * C_ + c) * 9 + kk];
    }
    __syncthreads();
#pragma unroll
    for (int kk = 0; kk < 9; ++kk) {
      int ky = kk / 3, kx = kk - (kk / 3) * 3;
      float v = xt[(py + ky) * 18 + (px + kx)];
#pragma unroll
      for (int ch = 0; ch < 27; ++ch)
        acc[ch] = fmaf(v, wl[ch * 9 + kk], acc[ch]);
    }
    __syncthreads();
  }
#pragma unroll
  for (int ch = 0; ch < 18; ++ch)
    off_ws[((b * 18 + ch) * H_ + y) * W_ + xw] = acc[ch];
#pragma unroll
  for (int j = 0; j < 9; ++j) {
    float v = acc[18 + j];
    msk_ws[((b * 9 + j) * H_ + y) * W_ + xw] = 1.0f / (1.0f + expf(-v));
  }
}

// ---------------------------------------------------------------------------
// Stage 2: fused bilinear-sample + GEMM with WMMA.
// WG (8 wave32) owns 32 pixels (b, y, w0..w0+31) x all 256 F.
// Per tap k (9 total):
//   - produce the FULL 32x256 f16 A tile for this tap into double-buffered
//     LDS (each thread: 2 pixels x 16 channels -> 4x ds_store_b128), 1 barrier
//   - then 8 chunks x { 4x ds_load_b128 (two A frags) + 4x global_load_b128
//     (two B frags, SHARED across both M-subtiles) + 4x v_wmma }
// x and Wt are L2-resident (32MB + 1.2MB << 192MB L2).
// ---------------------------------------------------------------------------
__global__ __launch_bounds__(256) void k_deform_wmma(const float* __restrict__ x,
                                                     const float* __restrict__ off_ws,
                                                     const float* __restrict__ msk_ws,
                                                     const _Float16* __restrict__ wt,
                                                     float* __restrict__ out) {
  __shared__ __align__(16) _Float16 As[2][32 * AST];

  int t   = threadIdx.x;
  int blk = blockIdx.x;
  int b   = blk >> 7;          // 128 tiles per image (64 rows x 2 segments)
  int rem = blk & 127;
  int y   = rem >> 1;
  int w0  = (rem & 1) << 5;

  int lane = t & 31;
  int wv   = t >> 5;
  int half = lane >> 4;
  int nl   = lane & 15;
  int fbase = wv * 32;
  int f0 = fbase + nl;
  int f1 = fbase + 16 + nl;

  int p0 = t & 15;        // this thread samples pixel rows p0 and p0+16
  int cg = (t >> 4) * 16; // ... for channels cg .. cg+15

  v8f acc00 = {}, acc01 = {};   // pixels w0..w0+15   x {f0, f1}
  v8f acc10 = {}, acc11 = {};   // pixels w0+16..w0+31 x {f0, f1}

  const float* xbase = x + (size_t)(b * C_) * (H_ * W_);
  int buf = 0;

  for (int k = 0; k < 9; ++k) {
    int kyi = k / 3 - 1;
    int kxi = k - (k / 3) * 3 - 1;

    // ---- bilinear params + A production for the 2 pixel rows ----
#pragma unroll
    for (int q = 0; q < 2; ++q) {
      int p  = p0 + q * 16;
      int xw = w0 + p;
      float dy = off_ws[((b * 18 + 2 * k + 0) * H_ + y) * W_ + xw];
      float dx = off_ws[((b * 18 + 2 * k + 1) * H_ + y) * W_ + xw];
      float mk = msk_ws[((b * 9 + k) * H_ + y) * W_ + xw];

      float fy = (float)y  + (float)kyi + dy;
      float fx = (float)xw + (float)kxi + dx;
      float y0f = floorf(fy), x0f = floorf(fx);
      float wy = fy - y0f, wxf = fx - x0f;
      int y0 = (int)y0f, x0 = (int)x0f;
      int y1 = y0 + 1,  x1 = x0 + 1;
      bool vy0 = (y0 >= 0) && (y0 < H_), vy1 = (y1 >= 0) && (y1 < H_);
      bool vx0 = (x0 >= 0) && (x0 < W_), vx1 = (x1 >= 0) && (x1 < W_);
      int y0c = min(max(y0, 0), H_ - 1), y1c = min(max(y1, 0), H_ - 1);
      int x0c = min(max(x0, 0), W_ - 1), x1c = min(max(x1, 0), W_ - 1);
      // fold per-corner validity and the modulation mask into the weights
      float w00 = (vy0 && vx0) ? (1.f - wy) * (1.f - wxf) * mk : 0.f;
      float w01 = (vy0 && vx1) ? (1.f - wy) * wxf * mk : 0.f;
      float w10 = (vy1 && vx0) ? wy * (1.f - wxf) * mk : 0.f;
      float w11 = (vy1 && vx1) ? wy * wxf * mk : 0.f;
      int i00 = y0c * W_ + x0c, i01 = y0c * W_ + x1c;
      int i10 = y1c * W_ + x0c, i11 = y1c * W_ + x1c;

#pragma unroll
      for (int s = 0; s < 2; ++s) {
        v8h pack;
#pragma unroll
        for (int u = 0; u < 8; ++u) {
          int c = cg + s * 8 + u;
          const float* xp = xbase + (size_t)c * (H_ * W_);
          float sm = xp[i00] * w00 + xp[i01] * w01 + xp[i10] * w10 + xp[i11] * w11;
          pack[u] = (_Float16)sm;
        }
        *reinterpret_cast<v8h*>(&As[buf][p * AST + cg + s * 8]) = pack;
      }
    }
    __syncthreads();

    // per-lane B row pointers for this tap: Wt[k][f][*] + K-group offset
    const _Float16* bp0 = wt + (size_t)(k * F_ + f0) * C_ + half * 8;
    const _Float16* bp1 = wt + (size_t)(k * F_ + f1) * C_ + half * 8;

    // ---- 8 WMMA chunks over the 256 channels of this tap ----
#pragma unroll
    for (int cc = 0; cc < 8; ++cc) {
      int c0 = cc * 32;
      // A fragments (ISA 16-bit A 16x32 layout): lane half 0 -> K {0-7,16-23},
      // half 1 -> K {8-15,24-31}; K maps to channel c0+K.
      v8h a0lo = *reinterpret_cast<const v8h*>(&As[buf][nl * AST + c0 + half * 8]);
      v8h a0hi = *reinterpret_cast<const v8h*>(&As[buf][nl * AST + c0 + 16 + half * 8]);
      v8h a1lo = *reinterpret_cast<const v8h*>(&As[buf][(16 + nl) * AST + c0 + half * 8]);
      v8h a1hi = *reinterpret_cast<const v8h*>(&As[buf][(16 + nl) * AST + c0 + 16 + half * 8]);
      v16h Af0 = __builtin_shufflevector(a0lo, a0hi, 0, 1, 2, 3, 4, 5, 6, 7,
                                         8, 9, 10, 11, 12, 13, 14, 15);
      v16h Af1 = __builtin_shufflevector(a1lo, a1hi, 0, 1, 2, 3, 4, 5, 6, 7,
                                         8, 9, 10, 11, 12, 13, 14, 15);

      // B fragments: two 8-half runs, 16 apart, straight from global (L2 hit),
      // shared by both M-subtiles.
      v8h b0lo = *reinterpret_cast<const v8h*>(bp0 + c0);
      v8h b0hi = *reinterpret_cast<const v8h*>(bp0 + c0 + 16);
      v8h b1lo = *reinterpret_cast<const v8h*>(bp1 + c0);
      v8h b1hi = *reinterpret_cast<const v8h*>(bp1 + c0 + 16);
      v16h Bf0 = __builtin_shufflevector(b0lo, b0hi, 0, 1, 2, 3, 4, 5, 6, 7,
                                         8, 9, 10, 11, 12, 13, 14, 15);
      v16h Bf1 = __builtin_shufflevector(b1lo, b1hi, 0, 1, 2, 3, 4, 5, 6, 7,
                                         8, 9, 10, 11, 12, 13, 14, 15);

      acc00 = __builtin_amdgcn_wmma_f32_16x16x32_f16(false, Af0, false, Bf0,
                                                     (short)0, acc00, false, false);
      acc01 = __builtin_amdgcn_wmma_f32_16x16x32_f16(false, Af0, false, Bf1,
                                                     (short)0, acc01, false, false);
      acc10 = __builtin_amdgcn_wmma_f32_16x16x32_f16(false, Af1, false, Bf0,
                                                     (short)0, acc10, false, false);
      acc11 = __builtin_amdgcn_wmma_f32_16x16x32_f16(false, Af1, false, Bf1,
                                                     (short)0, acc11, false, false);
    }
    buf ^= 1;   // double buffer: a single barrier per tap is sufficient
  }

  // ---- store D (lane: N = nl, VGPR r -> M = half*8 + r); rows of 8
  //      consecutive pixels -> global_store_b128 pairs ----
  size_t o0 = ((size_t)(b * F_ + f0) * H_ + y) * W_ + w0;
  size_t o1 = ((size_t)(b * F_ + f1) * H_ + y) * W_ + w0;
#pragma unroll
  for (int r = 0; r < 8; ++r) {
    int m = half * 8 + r;
    out[o0 + m]      = acc00[r];
    out[o1 + m]      = acc01[r];
    out[o0 + 16 + m] = acc10[r];
    out[o1 + 16 + m] = acc11[r];
  }
}

// ---------------------------------------------------------------------------
extern "C" void kernel_launch(void* const* d_in, const int* in_sizes, int n_in,
                              void* d_out, int out_size, void* d_ws, size_t ws_size,
                              hipStream_t stream) {
  const float* x  = (const float*)d_in[0];   // (8,256,64,64)
  const float* wo = (const float*)d_in[1];   // (18,256,3,3)
  const float* wm = (const float*)d_in[2];   // (9,256,3,3)
  const float* wd = (const float*)d_in[3];   // (256,256,3,3)
  float* out = (float*)d_out;                // (8,256,64,64)

  float* off_ws = (float*)d_ws;                                   // B*18*H*W f32
  float* msk_ws = off_ws + (size_t)B_ * 18 * H_ * W_;             // B*9*H*W  f32
  _Float16* wtp = (_Float16*)(msk_ws + (size_t)B_ * 9 * H_ * W_); // K*F*C    f16

  k_repack_w<<<(K_ * F_ * C_ + 255) / 256, 256, 0, stream>>>(wd, wtp);
  k_offset_mask<<<B_ * 16, 256, 0, stream>>>(x, wo, wm, off_ws, msk_ws);
  k_deform_wmma<<<B_ * H_ * (W_ / 32), 256, 0, stream>>>(x, off_ws, msk_ws, wtp, out);
}
